// Encoder_17076789969532
// MI455X (gfx1250) — compile-verified, matrix operand order checked by reading, not run on hardware
//
#include <hip/hip_runtime.h>
#include <hip/hip_bf16.h>
#include <math.h>

// ---------------------------------------------------------------------------
// MI455X (gfx1250) transformer encoder, bf16 WMMA everywhere.
//
// Shapes: N=4, T=2048, P=64, E=256, H=8, DH=32, FC=1024, NL=2, LOCAL=64.
// All GEMM K dims (64 / 256 / 1024) are multiples of 32 -> pure
// v_wmma_f32_16x16x32_bf16 pipelines, fp32 accumulation.
// Memory: ~300 MB activation round trips @ 23.3 TB/s is ~15us -> the chip is
// compute/latency bound here; staging is double-buffered through LDS so the
// WMMA pipes never wait on HBM.
// All flat indices fit in 32 bits (max ~8.4M elements) -> int indexing keeps
// address math in 32-bit VALU/SALU.
// ---------------------------------------------------------------------------

typedef __attribute__((ext_vector_type(16))) __bf16 v16bf;
typedef __attribute__((ext_vector_type(8)))  __bf16 v8bf;
typedef __attribute__((ext_vector_type(4)))  __bf16 v4bf;
typedef __attribute__((ext_vector_type(8)))  float  v8f;

#define Ncfg     4
#define Tcfg     2048
#define Pcfg     64
#define Ecfg     256
#define Hcfg     8
#define DHcfg    32
#define FCcfg    1024
#define NLcfg    2
#define LOCALcfg 64
#define TEcfg    (Tcfg * Ecfg)          // 524288
#define NTcfg    (Ncfg * Tcfg)          // 8192
#define NTEcfg   (Ncfg * TEcfg)         // 2097152  (fits int)
#define RBLK     64                     // reduction blocks per sample

__device__ __forceinline__ void pack4(v4bf& p, float4 a) {
    p[0] = (__bf16)a.x; p[1] = (__bf16)a.y; p[2] = (__bf16)a.z; p[3] = (__bf16)a.w;
}
__device__ __forceinline__ void pack8(v16bf& f, int base, float4 a, float4 b) {
    f[base + 0] = (__bf16)a.x; f[base + 1] = (__bf16)a.y;
    f[base + 2] = (__bf16)a.z; f[base + 3] = (__bf16)a.w;
    f[base + 4] = (__bf16)b.x; f[base + 5] = (__bf16)b.y;
    f[base + 6] = (__bf16)b.z; f[base + 7] = (__bf16)b.w;
}
// fragment gather: two 16B LDS chunks per lane (ds_load_b128)
__device__ __forceinline__ v16bf frag_from_lds(const __bf16* rowptr, int lh) {
    v8bf lo = *(const v8bf*)(rowptr + lh * 8);
    v8bf hi = *(const v8bf*)(rowptr + 16 + lh * 8);
    v16bf f;
#pragma unroll
    for (int e = 0; e < 8; ++e) { f[e] = lo[e]; f[8 + e] = hi[e]; }
    return f;
}

// ---------------------------------------------------------------------------
// Generic C[m][n] = sum_k A[m][k] * W[n][k] + bias[n], optional embed epilogue
// C = (acc + bias) * scale + pos[n*Tdim + (m & (Tdim-1))]   (Tdim power of 2).
// Block: 256 thr = 8 waves; block tile 128x64; each wave 32x32 (2x2 WMMA).
// LDS double-buffered; staging via global_load_b128 + ds_store_b64.
// Fragment layout per CDNA5 ISA 7.12.2:
//   A 16x32 bf16: lanes 0-15 row M=lane; element e -> K=(e<8?e:e+8)+(lane>=16?8:0)
//   B 32x16 bf16: col N = lane&15; same K mapping.
//   C/D 16x16 f32: row = v + 8*(lane>=16); col = lane&15.
// ---------------------------------------------------------------------------
__global__ __launch_bounds__(256) void k_gemm_bf16(
    const float* __restrict__ A, const float* __restrict__ W,
    const float* __restrict__ bias, float* __restrict__ C,
    int M, int N, int K,
    const float* __restrict__ pos, float scale, int Tdim)
{
    __shared__ __align__(16) __bf16 As[2][128 * 32];   // 2 x 8 KB
    __shared__ __align__(16) __bf16 Bs[2][64 * 32];    // 2 x 4 KB

    const int tid  = threadIdx.x;
    const int lane = tid & 31;
    const int wave = tid >> 5;
    const int lr   = lane & 15;
    const int lh   = lane >> 4;
    const int wm   = wave & 3;    // 4 waves along M
    const int wn   = wave >> 2;   // 2 waves along N
    const int bm   = blockIdx.y * 128;
    const int bn   = blockIdx.x * 64;

    v8f zacc = {};
    v8f acc[2][2];
    acc[0][0] = zacc; acc[0][1] = zacc; acc[1][0] = zacc; acc[1][1] = zacc;

    float4 ta[4], tb[2];
    // ---- issue tile loads for k-step `k0` into ta/tb (global_load_b128 x6) ----
    auto stage_load = [&](int k0) {
#pragma unroll
        for (int t = 0; t < 4; ++t) {
            int li = (t * 256 + tid) * 4;         // 0..4095, 16B chunks
            int r  = li >> 5, k = li & 31;
            ta[t] = *(const float4*)&A[(bm + r) * K + k0 + k];
        }
#pragma unroll
        for (int t = 0; t < 2; ++t) {
            int li = (t * 256 + tid) * 4;         // 0..2047
            int r  = li >> 5, k = li & 31;
            tb[t] = *(const float4*)&W[(bn + r) * K + k0 + k];
        }
    };
    // ---- convert + store staged tile into LDS buffer `buf` (ds_store_b64) ----
    auto stage_store = [&](int buf) {
#pragma unroll
        for (int t = 0; t < 4; ++t) {
            int li = (t * 256 + tid) * 4;
            v4bf p; pack4(p, ta[t]);
            *(v4bf*)&As[buf][li] = p;
        }
#pragma unroll
        for (int t = 0; t < 2; ++t) {
            int li = (t * 256 + tid) * 4;
            v4bf p; pack4(p, tb[t]);
            *(v4bf*)&Bs[buf][li] = p;
        }
    };

    const int nk = K / 32;
    stage_load(0);
    stage_store(0);
    __syncthreads();

    for (int kt = 0; kt < nk; ++kt) {
        const int cur = kt & 1;
        const bool has_next = (kt + 1) < nk;
        if (has_next) stage_load((kt + 1) * 32);   // loads in flight during WMMA

        v16bf af[2], bf[2];
#pragma unroll
        for (int i = 0; i < 2; ++i)
            af[i] = frag_from_lds(&As[cur][(wm * 32 + i * 16 + lr) * 32], lh);
#pragma unroll
        for (int j = 0; j < 2; ++j)
            bf[j] = frag_from_lds(&Bs[cur][(wn * 32 + j * 16 + lr) * 32], lh);

#pragma unroll
        for (int i = 0; i < 2; ++i)
#pragma unroll
            for (int j = 0; j < 2; ++j)
                acc[i][j] = __builtin_amdgcn_wmma_f32_16x16x32_bf16(
                    false, af[i], false, bf[j], (short)0, acc[i][j], false, false);

        if (has_next) stage_store(1 - cur);
        __syncthreads();
    }

    // ---- epilogue (pure 32-bit indexing; Tdim is a power of two) ----
    const int tmask = Tdim - 1;
#pragma unroll
    for (int i = 0; i < 2; ++i) {
#pragma unroll
        for (int j = 0; j < 2; ++j) {
            int col = bn + wn * 32 + j * 16 + lr;
            float bv = bias[col];
#pragma unroll
            for (int v = 0; v < 8; ++v) {
                int row = bm + wm * 32 + i * 16 + v + lh * 8;
                float r = acc[i][j][v] + bv;
                if (pos) r = r * scale + pos[col * Tdim + (row & tmask)];
                C[row * N + col] = r;
            }
        }
    }
}

// ---------------------------------------------------------------------------
// Banded local attention, one wave per (query-tile, head, batch).
// Band |t-s|<=64 -> <=144 keys per 16-row query tile -> 9 key tiles.
// Scores via WMMA (K=DH=32), softmax in LDS, P.V via two K=16-padded WMMAs
// per key tile with the V tile staged in LDS (bf16).
// qkv layout: [n][t][3E] with q|k|v at offsets 0|E|2E.
// ---------------------------------------------------------------------------
__global__ __launch_bounds__(32) void k_attn(const float* __restrict__ qkv,
                                             float* __restrict__ out)
{
    const int qt   = blockIdx.x * 16;
    const int h    = blockIdx.y;
    const int n    = blockIdx.z;
    const int lane = threadIdx.x;
    const int lr   = lane & 15;
    const int lh   = lane >> 4;

    __shared__ __align__(16) float  sc[16][144];
    __shared__ __align__(16) __bf16 pb[16][144];
    __shared__ __align__(16) __bf16 Vs[16][32];
    __shared__ float rsum[16];

    const int rowstride = 3 * Ecfg;

    // Q fragment (16x32): row = qt + lr; contiguous 8-float runs -> float4
    v16bf qf;
    {
        const float* qrow =
            qkv + (n * Tcfg + qt + lr) * rowstride + h * DHcfg + lh * 8;
        float4 a = *(const float4*)(qrow);
        float4 b = *(const float4*)(qrow + 4);
        float4 c = *(const float4*)(qrow + 16);
        float4 d = *(const float4*)(qrow + 20);
        pack8(qf, 0, a, b);
        pack8(qf, 8, c, d);
    }

    const float inv_sqrt_dh = 0.17677669529663687f; // 1/sqrt(32)

#pragma unroll 1
    for (int jt = 0; jt < 9; ++jt) {
        int st0 = qt - LOCALcfg + jt * 16;
        int s   = st0 + lr;                // key index for this lane's column
        bool ok = (s >= 0) && (s < Tcfg);
        v16bf kf;
        if (ok) {
            const float* krow =
                qkv + (n * Tcfg + s) * rowstride + Ecfg + h * DHcfg + lh * 8;
            float4 a = *(const float4*)(krow);
            float4 b = *(const float4*)(krow + 4);
            float4 c = *(const float4*)(krow + 16);
            float4 d = *(const float4*)(krow + 20);
            pack8(kf, 0, a, b);
            pack8(kf, 8, c, d);
        } else {
#pragma unroll
            for (int e = 0; e < 16; ++e) kf[e] = (__bf16)0.0f;
        }
        v8f s8 = {};
        s8 = __builtin_amdgcn_wmma_f32_16x16x32_bf16(
            false, qf, false, kf, (short)0, s8, false, false);
#pragma unroll
        for (int v = 0; v < 8; ++v) {
            int row = v + lh * 8;
            int tq  = qt + row;
            int dd  = tq - s; if (dd < 0) dd = -dd;
            bool bad = (!ok) || (dd > LOCALcfg);
            sc[row][jt * 16 + lr] = bad ? -3.0e38f : s8[v] * inv_sqrt_dh;
        }
    }
    __syncthreads();
    asm volatile("s_wait_dscnt 0" ::: "memory");

    // row-wise softmax (lanes 0..15, one row each)
    if (lane < 16) {
        float mx = -3.0e38f;
        for (int j = 0; j < 144; ++j) mx = fmaxf(mx, sc[lane][j]);
        float sum = 0.0f;
        for (int j = 0; j < 144; ++j) {
            float p = __expf(sc[lane][j] - mx);
            sum += p;
            pb[lane][j] = (__bf16)p;
        }
        rsum[lane] = 1.0f / sum;
    }
    __syncthreads();
    asm volatile("s_wait_dscnt 0" ::: "memory");

    // O = P.V  (two 16-col halves of DH=32), K=16 zero-padded to 32
    v8f zo = {};
    v8f o8[2]; o8[0] = zo; o8[1] = zo;
#pragma unroll 1
    for (int jt = 0; jt < 9; ++jt) {
        int st0 = qt - LOCALcfg + jt * 16;

        // ---- cooperative V-tile stage: lane -> (row = lane>>1, 16-col half) ----
        {
            int r = lane >> 1;
            int c = (lane & 1) * 16;
            int s = st0 + r;
            v4bf z4 = {};
            if (s >= 0 && s < Tcfg) {
                const float* vrow =
                    qkv + (n * Tcfg + s) * rowstride + 2 * Ecfg + h * DHcfg + c;
                float4 a = *(const float4*)(vrow);
                float4 b = *(const float4*)(vrow + 4);
                float4 cc = *(const float4*)(vrow + 8);
                float4 d = *(const float4*)(vrow + 12);
                v4bf p0, p1, p2, p3;
                pack4(p0, a); pack4(p1, b); pack4(p2, cc); pack4(p3, d);
                *(v4bf*)&Vs[r][c + 0]  = p0;
                *(v4bf*)&Vs[r][c + 4]  = p1;
                *(v4bf*)&Vs[r][c + 8]  = p2;
                *(v4bf*)&Vs[r][c + 12] = p3;
            } else {
                *(v4bf*)&Vs[r][c + 0]  = z4;
                *(v4bf*)&Vs[r][c + 4]  = z4;
                *(v4bf*)&Vs[r][c + 8]  = z4;
                *(v4bf*)&Vs[r][c + 12] = z4;
            }
        }
        asm volatile("s_wait_dscnt 0" ::: "memory");

        // A frag: probabilities, K = key-in-tile (0..15 valid, upper half zero)
        v16bf pf;
        {
            v8bf pv = *(const v8bf*)&pb[lr][jt * 16 + lh * 8];
#pragma unroll
            for (int e = 0; e < 8; ++e) { pf[e] = pv[e]; pf[8 + e] = (__bf16)0.0f; }
        }
#pragma unroll
        for (int dhalf = 0; dhalf < 2; ++dhalf) {
            int d = dhalf * 16 + lr;       // output column = value dim
            v16bf vf;
#pragma unroll
            for (int e = 0; e < 8; ++e) {
                vf[e]     = Vs[lh * 8 + e][d];
                vf[8 + e] = (__bf16)0.0f;
            }
            o8[dhalf] = __builtin_amdgcn_wmma_f32_16x16x32_bf16(
                false, pf, false, vf, (short)0, o8[dhalf], false, false);
        }
    }

    asm volatile("s_wait_dscnt 0" ::: "memory");
#pragma unroll
    for (int dhalf = 0; dhalf < 2; ++dhalf) {
#pragma unroll
        for (int v = 0; v < 8; ++v) {
            int row = v + lh * 8;
            float o = o8[dhalf][v] * rsum[row];
            out[(n * Tcfg + qt + row) * Ecfg + h * DHcfg + dhalf * 16 + lr] = o;
        }
    }
}

// ---------------------------------------------------------------------------
// r = a + b; per-block partial (sum, sumsq) for per-sample LayerNorm stats.
// grid = (RBLK, N); float4 throughout.
// ---------------------------------------------------------------------------
__global__ __launch_bounds__(256) void k_resid_reduce(
    const float* __restrict__ a, const float* __restrict__ b,
    float* __restrict__ r, float* __restrict__ partial)
{
    const int n      = blockIdx.y;
    const int chunk4 = (TEcfg / RBLK) / 4;                 // 2048 float4
    const int base4  = (n * TEcfg + blockIdx.x * (TEcfg / RBLK)) / 4;
    const float4* a4 = (const float4*)a;
    const float4* b4 = (const float4*)b;
    float4*       r4 = (float4*)r;
    float s = 0.0f, s2 = 0.0f;
    for (int i = threadIdx.x; i < chunk4; i += 256) {
        float4 va = a4[base4 + i];
        float4 vb = b4[base4 + i];
        float4 v;
        v.x = va.x + vb.x; v.y = va.y + vb.y;
        v.z = va.z + vb.z; v.w = va.w + vb.w;
        r4[base4 + i] = v;
        s  += v.x + v.y + v.z + v.w;
        s2 += v.x * v.x + v.y * v.y + v.z * v.z + v.w * v.w;
    }
    __shared__ float r1[256], r2[256];
    r1[threadIdx.x] = s; r2[threadIdx.x] = s2;
    __syncthreads();
    for (int off = 128; off > 0; off >>= 1) {
        if (threadIdx.x < off) {
            r1[threadIdx.x] += r1[threadIdx.x + off];
            r2[threadIdx.x] += r2[threadIdx.x + off];
        }
        __syncthreads();
    }
    if (threadIdx.x == 0) {
        partial[(n * RBLK + blockIdx.x) * 2 + 0] = r1[0];
        partial[(n * RBLK + blockIdx.x) * 2 + 1] = r2[0];
    }
}

// one block per sample: fold RBLK partials -> (mu, rsigma)
__global__ __launch_bounds__(64) void k_ln_stats(const float* __restrict__ partial,
                                                 float* __restrict__ stats)
{
    const int n = blockIdx.x, t = threadIdx.x;
    __shared__ float s1[64], s2[64];
    s1[t] = partial[(n * RBLK + t) * 2 + 0];
    s2[t] = partial[(n * RBLK + t) * 2 + 1];
    __syncthreads();
    for (int off = 32; off > 0; off >>= 1) {
        if (t < off) { s1[t] += s1[t + off]; s2[t] += s2[t + off]; }
        __syncthreads();
    }
    if (t == 0) {
        float mu  = s1[0] / (float)TEcfg;
        float var = s2[0] / (float)TEcfg - mu * mu;
        stats[n * 2 + 0] = mu;
        stats[n * 2 + 1] = rsqrtf(var + 1e-5f);
    }
}

// out[i] = (r[i]-mu)*rsig * g[i%TE] + b[i%TE]   (float4, 32-bit indexing)
__global__ __launch_bounds__(256) void k_ln_apply(
    const float* __restrict__ r, const float* __restrict__ g,
    const float* __restrict__ b, const float* __restrict__ stats,
    float* __restrict__ out)
{
    const float4* r4 = (const float4*)r;
    const float4* g4 = (const float4*)g;
    const float4* b4 = (const float4*)b;
    float4*       o4 = (float4*)out;
    const int total4 = NTEcfg / 4;       // 524288
    const int te4    = TEcfg / 4;        // 131072 (power of two)
    for (int i = blockIdx.x * 256 + threadIdx.x; i < total4;
         i += gridDim.x * 256) {
        int n = i >> 17;                 // i / te4
        int j = i & (te4 - 1);           // i % te4
        float mu = stats[n * 2], rs = stats[n * 2 + 1];
        float4 rv = r4[i], gv = g4[j], bv = b4[j];
        float4 ov;
        ov.x = (rv.x - mu) * rs * gv.x + bv.x;
        ov.y = (rv.y - mu) * rs * gv.y + bv.y;
        ov.z = (rv.z - mu) * rs * gv.z + bv.z;
        ov.w = (rv.w - mu) * rs * gv.w + bv.w;
        o4[i] = ov;
    }
}

// ---------------------------------------------------------------------------
extern "C" void kernel_launch(void* const* d_in, const int* in_sizes, int n_in,
                              void* d_out, int out_size, void* d_ws, size_t ws_size,
                              hipStream_t stream)
{
    (void)in_sizes; (void)n_in; (void)out_size; (void)ws_size;

    const float* input = (const float*)d_in[0];
    const float* w_emb = (const float*)d_in[1];
    const float* b_emb = (const float*)d_in[2];
    const float* pos   = (const float*)d_in[3];   // (E, T)
    const float* wqkv  = (const float*)d_in[4];   // (NL, 3E, E)
    const float* bqkv  = (const float*)d_in[5];   // (NL, 3E)
    const float* wo    = (const float*)d_in[6];   // (NL, E, E)
    const float* bo    = (const float*)d_in[7];
    const float* ln1g  = (const float*)d_in[8];   // (NL, T, E)
    const float* ln1b  = (const float*)d_in[9];
    const float* ln2g  = (const float*)d_in[10];
    const float* ln2b  = (const float*)d_in[11];
    const float* w1    = (const float*)d_in[12];  // (NL, FC, E)
    const float* b1    = (const float*)d_in[13];
    const float* w2    = (const float*)d_in[14];  // (NL, E, FC)
    const float* b2    = (const float*)d_in[15];

    // workspace layout (floats) -- ~101 MB total
    float* ws      = (float*)d_ws;
    float* x       = ws;                                // NTE
    float* qkv     = x      + NTEcfg;                   // NT * 3E
    float* attno   = qkv    + NTcfg * 3 * Ecfg;         // NTE
    float* obuf    = attno  + NTEcfg;                   // NTE (also reused as f2)
    float* rbuf    = obuf   + NTEcfg;                   // NTE
    float* hbuf    = rbuf   + NTEcfg;                   // NTE
    float* f1      = hbuf   + NTEcfg;                   // NT * FC
    float* partial = f1     + NTcfg * FCcfg;            // N*RBLK*2
    float* stats   = partial + Ncfg * RBLK * 2;         // N*2

    const dim3 blk256(256), blk32(32), blk64(64);

    // embedding: x = (frames @ w_emb^T + b_emb) * sqrt(E) + pos
    k_gemm_bf16<<<dim3(Ecfg / 64, NTcfg / 128), blk256, 0, stream>>>(
        input, w_emb, b_emb, x, NTcfg, Ecfg, Pcfg, pos, 16.0f, Tcfg);

    for (int l = 0; l < NLcfg; ++l) {
        const float* wqkv_l = wqkv + l * 3 * Ecfg * Ecfg;
        const float* bqkv_l = bqkv + l * 3 * Ecfg;
        const float* wo_l   = wo   + l * Ecfg * Ecfg;
        const float* bo_l   = bo   + l * Ecfg;
        const float* w1_l   = w1   + l * FCcfg * Ecfg;
        const float* b1_l   = b1   + l * FCcfg;
        const float* w2_l   = w2   + l * Ecfg * FCcfg;
        const float* b2_l   = b2   + l * Ecfg;

        // qkv = x @ Wqkv^T + b            (8192 x 768, K=256)
        k_gemm_bf16<<<dim3(3 * Ecfg / 64, NTcfg / 128), blk256, 0, stream>>>(
            x, wqkv_l, bqkv_l, qkv, NTcfg, 3 * Ecfg, Ecfg, nullptr, 1.0f, Tcfg);

        // banded attention -> attno
        k_attn<<<dim3(Tcfg / 16, Hcfg, Ncfg), blk32, 0, stream>>>(qkv, attno);

        // o = attno @ Wo^T + bo           (8192 x 256, K=256)
        k_gemm_bf16<<<dim3(Ecfg / 64, NTcfg / 128), blk256, 0, stream>>>(
            attno, wo_l, bo_l, obuf, NTcfg, Ecfg, Ecfg, nullptr, 1.0f, Tcfg);

        // h = LN(x + o)
        k_resid_reduce<<<dim3(RBLK, Ncfg), blk256, 0, stream>>>(x, obuf, rbuf, partial);
        k_ln_stats<<<dim3(Ncfg), blk64, 0, stream>>>(partial, stats);
        k_ln_apply<<<dim3(2048), blk256, 0, stream>>>(
            rbuf, ln1g + l * TEcfg, ln1b + l * TEcfg, stats, hbuf);

        // f = (h @ W1^T + b1) @ W2^T + b2
        k_gemm_bf16<<<dim3(FCcfg / 64, NTcfg / 128), blk256, 0, stream>>>(
            hbuf, w1_l, b1_l, f1, NTcfg, FCcfg, Ecfg, nullptr, 1.0f, Tcfg);
        k_gemm_bf16<<<dim3(Ecfg / 64, NTcfg / 128), blk256, 0, stream>>>(
            f1, w2_l, b2_l, obuf, NTcfg, Ecfg, FCcfg, nullptr, 1.0f, Tcfg);

        // x = LN(h + f)   (last layer writes straight to d_out)
        k_resid_reduce<<<dim3(RBLK, Ncfg), blk256, 0, stream>>>(hbuf, obuf, rbuf, partial);
        k_ln_stats<<<dim3(Ncfg), blk64, 0, stream>>>(partial, stats);
        float* dst = (l == NLcfg - 1) ? (float*)d_out : x;
        k_ln_apply<<<dim3(2048), blk256, 0, stream>>>(
            rbuf, ln2g + l * TEcfg, ln2b + l * TEcfg, stats, dst);
    }
}
